// PointNet2Backbone_85899345944
// MI455X (gfx1250) — matrix operand order repaired and checked by reading, not compiled, not run on hardware
//
#include <hip/hip_runtime.h>
#include <hip/hip_bf16.h>

// PointNet++ MSG backbone for MI455X (gfx1250, wave32).
//   init -> prep weights (f32->f16, K padded to 32) -> ball query x3
//   -> per scale: [WMMA GEMM(+BN-stat atomics) -> finalize stats] x3 -> maxpool
//   -> projection GEMM -> finalize -> final BN+ReLU f32 output.
// Intermediates f16 (memory-bound @ 23.3 TB/s); matmuls v_wmma_f32_16x16x32_f16.
// GEMM steady state is LDS-only (A panel + B slice both staged), vectorized
// b128 staging loads, register-pre-reduced BN statistics.
// Workspace requirement: ~250 MB.

typedef _Float16 v16h __attribute__((ext_vector_type(16)));
typedef _Float16 v8h  __attribute__((ext_vector_type(8)));
typedef float    v8f  __attribute__((ext_vector_type(8)));
typedef float    v4f  __attribute__((ext_vector_type(4)));

#define NPTS 2048
#define NBATCH 4

// ---------------------------------------------------------------- init
__global__ void init_kernel(float* stats, float* coef) {
  int g = blockIdx.x * 256 + threadIdx.x;
  if (g < 10 * 512) stats[g] = 0.0f;          // zero all per-layer (sum,sumsq)
  if (g < 512) {                              // identity coefs (layer slot 10)
    float* ic = coef + 10 * 512;
    ic[g] = (g < 256) ? 1.0f : 0.0f;
  }
}

// ------------------------------------------------------- weight prep (f16)
__global__ void prep_w_kernel(const float* __restrict__ W, _Float16* __restrict__ W16,
                              int co, int ci, int ci_pad) {
  int g = blockIdx.x * 256 + threadIdx.x;
  if (g >= co * ci_pad) return;
  int o = g / ci_pad, c = g - o * ci_pad;
  W16[g] = (_Float16)(c < ci ? W[o * ci + c] : 0.0f);
}

// ---------------------------------------------------------------- ball query
// Reference sorts (index-if-in-radius else N) and keeps k smallest == first k
// in-radius indices in scan order, padded with the first hit (self always hits).
__global__ void ball_query_kernel(const float* __restrict__ xyz, int* __restrict__ idx,
                                  float r2, int k) {
  __shared__ float sx[NPTS * 3];
  int b = blockIdx.y;
  const float* bx = xyz + (size_t)b * NPTS * 3;
  for (int e = threadIdx.x; e < NPTS * 3; e += 256) sx[e] = bx[e];
  __syncthreads();
  int n = blockIdx.x * 256 + threadIdx.x;
  float cx = sx[n * 3], cy = sx[n * 3 + 1], cz = sx[n * 3 + 2];
  int* out = idx + (size_t)(b * NPTS + n) * k;
  int cnt = 0;
  for (int j = 0; j < NPTS && cnt < k; ++j) {
    float dx = sx[j * 3] - cx, dy = sx[j * 3 + 1] - cy, dz = sx[j * 3 + 2] - cz;
    if (dx * dx + dy * dy + dz * dz < r2) out[cnt++] = j;
  }
  int first = out[0];
  for (; cnt < k; ++cnt) out[cnt] = first;
}

// ------------------------------------------------ fused GEMM + BN statistics
// Y[M x co] = A[M x ci_pad] * W16^T.  Block: 256 thr / 8 waves; block tile =
// 128 rows x 32 cols (grid.y slices co).  A panel (128 x ci_pad, producer
// BN+ReLU fused, or gathered rel-coords) AND the shared B slice (32 x ci_pad)
// are staged in LDS; inner loop is ds_load_b128 + v_wmma only.
__global__ void gemm_bn_kernel(const _Float16* __restrict__ Asrc,  // mode1: M x ci_pad f16
                               const float* __restrict__ xyz,      // mode0
                               const int* __restrict__ idx,        // mode0
                               const float* __restrict__ coef,     // [a(256), b(256)] input xform
                               const _Float16* __restrict__ W16,   // co x ci_pad
                               _Float16* __restrict__ Y,           // M x co
                               float* __restrict__ stat,           // [sum(256), sumsq(256)]
                               int ci_pad, int co, int mode, int kNb) {
  extern __shared__ char smem_raw[];
  _Float16* As = (_Float16*)smem_raw;                               // 128 x ci_pad
  _Float16* Bs = As + (size_t)128 * ci_pad;                         // 32 x ci_pad
  float* s_sum  = (float*)(Bs + (size_t)32 * ci_pad);               // 32
  float* s_sq   = s_sum + 32;                                       // 32
  float* s_coef = s_sq + 32;                                        // 512

  const int tid = threadIdx.x;
  const int rowbase = blockIdx.x * 128;
  const int colbase = blockIdx.y * 32;

  // phase 0: coefs -> LDS, zero stats
  for (int e = tid; e < 512; e += 256) s_coef[e] = coef[e];
  if (tid < 64) s_sum[tid] = 0.0f;
  __syncthreads();

  // phase 1: stage B slice (shared by all 8 waves) and A panel
  {
    int bch = (32 * ci_pad) / 8;
    for (int e = tid; e < bch; e += 256) {
      int h0 = e * 8;
      int r = h0 / ci_pad, c = h0 - r * ci_pad;
      *(v8h*)(Bs + h0) = *(const v8h*)(W16 + (size_t)(colbase + r) * ci_pad + c);
    }
  }
  if (mode == 0) {  // gather: grouped = xyz[nb] - xyz[center]; ci_pad == 32
    if (tid < 128) {
      int m = rowbase + tid;
      int p = m / kNb;
      int b = p >> 11, n = p & (NPTS - 1);
      int nb = idx[m];
      const float* cp = xyz + (size_t)(b * NPTS + n) * 3;
      const float* np = xyz + (size_t)(b * NPTS + nb) * 3;
      v8h z = {};
      v8h h = z;
      h[0] = (_Float16)(np[0] - cp[0]);
      h[1] = (_Float16)(np[1] - cp[1]);
      h[2] = (_Float16)(np[2] - cp[2]);
      _Float16* dst = As + (size_t)tid * 32;
      *(v8h*)(dst) = h; *(v8h*)(dst + 8) = z; *(v8h*)(dst + 16) = z; *(v8h*)(dst + 24) = z;
    }
  } else {          // producer BN+ReLU fused into staging; ci_pad == src width
    int chunks = (128 * ci_pad) / 8;
    for (int e = tid; e < chunks; e += 256) {
      int h0 = e * 8;
      int r = h0 / ci_pad, c = h0 - r * ci_pad;   // c multiple of 8
      v8h x = *(const v8h*)(Asrc + (size_t)(rowbase + r) * ci_pad + c);
      v8h y;
      #pragma unroll
      for (int u = 0; u < 8; ++u) {
        float v = (float)x[u] * s_coef[c + u] + s_coef[256 + c + u];
        y[u] = (_Float16)(v > 0.0f ? v : 0.0f);
      }
      *(v8h*)(As + h0) = y;
    }
  }
  __syncthreads();

  const int w = tid >> 5, lane = tid & 31;
  const int lrow = lane & 15, lhalf = lane >> 4;

  v8f acc0 = {}, acc1 = {};
  const _Float16* Arow  = As + (size_t)(w * 16 + lrow) * ci_pad + lhalf * 8;
  const _Float16* Brow0 = Bs + (size_t)lrow * ci_pad + lhalf * 8;
  const _Float16* Brow1 = Bs + (size_t)(16 + lrow) * ci_pad + lhalf * 8;

  for (int kk = 0; kk < ci_pad; kk += 32) {
    // A frag (ISA 16-bit A 16x32 layout): halves 0..7 = K k0..k0+7, 8..15 = K k0+16..k0+23
    v8h lo = *(const v8h*)(Arow + kk);
    v8h hi = *(const v8h*)(Arow + kk + 16);
    v16h af = __builtin_shufflevector(lo, hi, 0, 1, 2, 3, 4, 5, 6, 7,
                                      8, 9, 10, 11, 12, 13, 14, 15);
    v16h bf0 = __builtin_shufflevector(*(const v8h*)(Brow0 + kk), *(const v8h*)(Brow0 + kk + 16),
                                       0, 1, 2, 3, 4, 5, 6, 7, 8, 9, 10, 11, 12, 13, 14, 15);
    v16h bf1 = __builtin_shufflevector(*(const v8h*)(Brow1 + kk), *(const v8h*)(Brow1 + kk + 16),
                                       0, 1, 2, 3, 4, 5, 6, 7, 8, 9, 10, 11, 12, 13, 14, 15);
    acc0 = __builtin_amdgcn_wmma_f32_16x16x32_f16(false, af, false, bf0, (short)0,
                                                  acc0, false, false);
    acc1 = __builtin_amdgcn_wmma_f32_16x16x32_f16(false, af, false, bf1, (short)0,
                                                  acc1, false, false);
  }

  // C/D layout: lane = n + 16*(m/8), vgpr i -> m = i + 8*(lane/16).
  // Each accumulator's 8 values share one column -> register pre-reduce stats.
  const int orow = rowbase + w * 16 + lhalf * 8;
  const int col0 = colbase + lrow;
  float sum0 = 0.0f, sq0 = 0.0f, sum1 = 0.0f, sq1 = 0.0f;
  #pragma unroll
  for (int i = 0; i < 8; ++i) {
    float v0 = acc0[i], v1 = acc1[i];
    size_t r = (size_t)(orow + i) * co;
    Y[r + col0]      = (_Float16)v0;
    Y[r + col0 + 16] = (_Float16)v1;
    sum0 += v0; sq0 += v0 * v0;
    sum1 += v1; sq1 += v1 * v1;
  }
  atomicAdd(&s_sum[lrow], sum0);
  atomicAdd(&s_sq[lrow],  sq0);
  atomicAdd(&s_sum[lrow + 16], sum1);
  atomicAdd(&s_sq[lrow + 16],  sq1);
  __syncthreads();
  if (tid < 32) {
    atomicAdd(&stat[colbase + tid], s_sum[tid]);
    atomicAdd(&stat[256 + colbase + tid], s_sq[tid]);
  }
}

// -------------------------------------------------------- BN stat finalize
__global__ void finalize_stats(const float* __restrict__ stat,
                               const float* __restrict__ gamma,
                               const float* __restrict__ beta,
                               float* __restrict__ coef, int co, float invM) {
  int c = threadIdx.x;
  if (c < co) {
    float mean = stat[c] * invM;
    float var  = stat[256 + c] * invM - mean * mean;
    float a = gamma[c] * rsqrtf(var + 1e-5f);
    coef[c] = a;
    coef[256 + c] = beta[c] - mean * a;
  }
}

// ------------------------------------------- BN+ReLU + max over neighbors
// One thread per (point, 8-channel chunk); v8h loads down the k axis.
__global__ void maxpool_kernel(const _Float16* __restrict__ Y, const float* __restrict__ coef,
                               _Float16* __restrict__ fused, int co, int k, int off) {
  int cw = co >> 3;  // 8-channel chunks per point
  int g = blockIdx.x * 256 + threadIdx.x;
  if (g >= NBATCH * NPTS * cw) return;
  int p = g / cw, cb = (g - p * cw) * 8;
  float a[8], bb[8], m[8];
  #pragma unroll
  for (int u = 0; u < 8; ++u) { a[u] = coef[cb + u]; bb[u] = coef[256 + cb + u]; m[u] = 0.0f; }
  const _Float16* yp = Y + (size_t)p * k * co + cb;
  for (int j = 0; j < k; ++j) {
    v8h x = *(const v8h*)(yp + (size_t)j * co);
    #pragma unroll
    for (int u = 0; u < 8; ++u) {
      float v = (float)x[u] * a[u] + bb[u];
      if (v > m[u]) m[u] = v;   // relu outputs >= 0, 0-init is the relu floor
    }
  }
  v8h o;
  #pragma unroll
  for (int u = 0; u < 8; ++u) o[u] = (_Float16)m[u];
  *(v8h*)(fused + (size_t)p * 320 + off + cb) = o;
}

// --------------------------------------------------------- final output
__global__ void final_out_kernel(const _Float16* __restrict__ Yp,
                                 const float* __restrict__ coef, float* __restrict__ out) {
  int g = blockIdx.x * 256 + threadIdx.x;   // 8192*256/8 chunks
  int base = g * 8;
  int c = base & 255;
  v8h x = *(const v8h*)(Yp + base);
  v4f o0, o1;
  #pragma unroll
  for (int u = 0; u < 4; ++u) {
    float v = (float)x[u] * coef[c + u] + coef[256 + c + u];
    o0[u] = v > 0.0f ? v : 0.0f;
  }
  #pragma unroll
  for (int u = 0; u < 4; ++u) {
    float v = (float)x[4 + u] * coef[c + 4 + u] + coef[256 + c + 4 + u];
    o1[u] = v > 0.0f ? v : 0.0f;
  }
  *(v4f*)(out + base) = o0;
  *(v4f*)(out + base + 4) = o1;
}

// ================================================================ launch
extern "C" void kernel_launch(void* const* d_in, const int* in_sizes, int n_in,
                              void* d_out, int out_size, void* d_ws, size_t ws_size,
                              hipStream_t stream) {
  const float* xyz = (const float*)d_in[0];

  // scale specs
  static const int   KNB[3]   = {16, 32, 64};
  static const float R2[3]    = {0.1f * 0.1f, 0.2f * 0.2f, 0.4f * 0.4f};
  static const int   CH[3][4] = {{3, 32, 32, 64}, {3, 64, 64, 128}, {3, 64, 96, 128}};

  // workspace carve-up
  char* ws = (char*)d_ws;
  size_t off = 0;
  auto alloc = [&](size_t bytes) { size_t o = off; off = (off + bytes + 255) & ~(size_t)255; return o; };
  size_t o_idx[3];
  for (int s = 0; s < 3; ++s) o_idx[s] = alloc((size_t)NBATCH * NPTS * KNB[s] * 4);
  size_t o_w16[10];
  for (int s = 0; s < 3; ++s)
    for (int l = 0; l < 3; ++l) {
      int ci = CH[s][l], co = CH[s][l + 1];
      int cip = (ci + 31) & ~31;
      o_w16[s * 3 + l] = alloc((size_t)co * cip * 2);
    }
  o_w16[9] = alloc((size_t)256 * 320 * 2);
  size_t o_stat = alloc(10 * 512 * 4);
  size_t o_coef = alloc(11 * 512 * 4);
  size_t o_fused = alloc((size_t)NBATCH * NPTS * 320 * 2);
  size_t o_yp    = alloc((size_t)NBATCH * NPTS * 256 * 2);
  size_t o_bufA  = alloc((size_t)NBATCH * NPTS * 64 * 96 * 2);   // <=96-wide outputs
  size_t o_bufB  = alloc((size_t)NBATCH * NPTS * 64 * 128 * 2);  // <=128-wide outputs
  (void)ws_size; (void)n_in; (void)in_sizes; (void)out_size;

  float* stats = (float*)(ws + o_stat);
  float* coef  = (float*)(ws + o_coef);
  float* coefIdent = coef + 10 * 512;

  // 1) init stats + identity coefs
  init_kernel<<<20, 256, 0, stream>>>(stats, coef);

  // 2) weight prep
  for (int s = 0; s < 3; ++s)
    for (int l = 0; l < 3; ++l) {
      int ci = CH[s][l], co = CH[s][l + 1];
      int cip = (ci + 31) & ~31;
      const float* W = (const float*)d_in[1 + s * 9 + l * 3];
      prep_w_kernel<<<(co * cip + 255) / 256, 256, 0, stream>>>(
          W, (_Float16*)(ws + o_w16[s * 3 + l]), co, ci, cip);
    }
  prep_w_kernel<<<(256 * 320 + 255) / 256, 256, 0, stream>>>(
      (const float*)d_in[28], (_Float16*)(ws + o_w16[9]), 256, 320, 320);

  // 3) ball query per scale
  for (int s = 0; s < 3; ++s)
    ball_query_kernel<<<dim3(NPTS / 256, NBATCH), 256, 0, stream>>>(
        xyz, (int*)(ws + o_idx[s]), R2[s], KNB[s]);

  // 4) per-scale MLP pipeline (L1 out->bufB, L2->bufA, L3->bufB)
  int fused_off = 0;
  for (int s = 0; s < 3; ++s) {
    int M = NBATCH * NPTS * KNB[s];
    size_t o_src = 0, o_dst = o_bufB;
    for (int l = 0; l < 3; ++l) {
      int ci = CH[s][l], co = CH[s][l + 1];
      int cip = (ci + 31) & ~31;
      int layer = s * 3 + l;
      int mode = (l == 0) ? 0 : 1;
      const float* in_coef = (l == 0) ? coefIdent : (coef + (size_t)(layer - 1) * 512);
      size_t shmem = (size_t)160 * cip * 2 + 64 * 4 + 512 * 4;
      gemm_bn_kernel<<<dim3(M / 128, co / 32), 256, shmem, stream>>>(
          (const _Float16*)(ws + o_src), xyz, (const int*)(ws + o_idx[s]),
          in_coef, (const _Float16*)(ws + o_w16[layer]),
          (_Float16*)(ws + o_dst), stats + (size_t)layer * 512,
          cip, co, mode, KNB[s]);
      finalize_stats<<<1, 256, 0, stream>>>(
          stats + (size_t)layer * 512, (const float*)d_in[1 + s * 9 + l * 3 + 1],
          (const float*)d_in[1 + s * 9 + l * 3 + 2], coef + (size_t)layer * 512,
          co, 1.0f / (float)M);
      o_src = o_dst;
      o_dst = (o_dst == o_bufB) ? o_bufA : o_bufB;
    }
    int co3 = CH[s][3];
    maxpool_kernel<<<(NBATCH * NPTS * (co3 >> 3) + 255) / 256, 256, 0, stream>>>(
        (const _Float16*)(ws + o_bufB), coef + (size_t)(s * 3 + 2) * 512,
        (_Float16*)(ws + o_fused), co3, KNB[s], fused_off);
    fused_off += co3;
  }

  // 5) projection 320 -> 256 (fused is post-activation; identity input coefs)
  {
    int M = NBATCH * NPTS;
    size_t shmem = (size_t)160 * 320 * 2 + 64 * 4 + 512 * 4;
    gemm_bn_kernel<<<dim3(M / 128, 256 / 32), 256, shmem, stream>>>(
        (const _Float16*)(ws + o_fused), xyz, nullptr, coefIdent,
        (const _Float16*)(ws + o_w16[9]), (_Float16*)(ws + o_yp),
        stats + 9 * 512, 320, 256, 1, 1);
    finalize_stats<<<1, 256, 0, stream>>>(stats + 9 * 512, (const float*)d_in[29],
                                          (const float*)d_in[30], coef + 9 * 512,
                                          256, 1.0f / (float)M);
  }

  // 6) final BN+ReLU -> f32 output
  final_out_kernel<<<(NBATCH * NPTS * 256 / 8 + 255) / 256, 256, 0, stream>>>(
      (const _Float16*)(ws + o_yp), coef + 9 * 512, (float*)d_out);
}
